// NeuralODE_18227841204740
// MI455X (gfx1250) — compile-verified, matrix operand order checked by reading, not run on hardware
//
#include <hip/hip_runtime.h>
#include <math.h>

typedef __attribute__((ext_vector_type(16))) _Float16 v16h;
typedef __attribute__((ext_vector_type(8)))  _Float16 v8h;
typedef __attribute__((ext_vector_type(8)))  float    v8f;

#define DIM   64
#define HID   256
#define TLEN  21
#define BSZ   65536
#define ROWS_PER_BLOCK 128   // 8 waves * 16 rows
#define THREADS 256

// LDS byte offsets
#define OFF_W1   0           // 64x256 f16, swizzled B-frag layout: 32 KB
#define OFF_W2   32768       // 256x64 f16, swizzled B-frag layout: 32 KB
#define OFF_Z    65536       // per-wave 16x64 f16 row-major: 2 KB * 8 = 16 KB
#define OFF_H    81920       // per-wave 16x256 f16 row-major: 8 KB * 8 = 64 KB
#define OFF_B1   147456      // 256 f32 bias1: 1 KB
#define OFF_B2   148480      // 64 f32 bias2: 256 B
#define SMEM_BYTES 148736

// fast tanh: 1 - 2/(exp2(2*log2e*x)+1); v_exp/v_rcp are TRANS ops that
// co-execute with WMMA. Saturates correctly for |x| large, propagates NaN.
__device__ __forceinline__ float fast_tanh(float x) {
    float e = __builtin_amdgcn_exp2f(x * 2.8853900817779268f);
    return 1.0f - 2.0f * __builtin_amdgcn_rcpf(e + 1.0f);
}

// ---- WMMA fragment loaders -------------------------------------------------
// A operand (16x32 f16, MxK): lane holds row m = lane%16; halves of the v16h
// cover k = 32*kt + 8*half + {0..7} and k = 32*kt + 8*half + 16 + {0..7},
// which are two contiguous 16B runs in a row-major f16 tile.
__device__ __forceinline__ v16h load_afrag(const _Float16* rowbase, int rowHalves,
                                           int m, int half, int kt) {
    const _Float16* p = rowbase + m * rowHalves + kt * 32 + half * 8;
    v8h lo = *(const v8h*)(p);
    v8h hi = *(const v8h*)(p + 16);
    v16h a;
#pragma unroll
    for (int i = 0; i < 8; ++i) { a[i] = lo[i]; a[i + 8] = hi[i]; }
    return a;
}

// B operand: pre-swizzled so each lane's 16 halves are contiguous (32B).
__device__ __forceinline__ v16h load_bfrag(const _Float16* fragbase, int lane) {
    return *(const v16h*)(fragbase + lane * 16);
}

// ---- stage z (or z + coef*k) into wave-private LDS tile (16x64 f16) --------
__device__ __forceinline__ void stage_z(_Float16* zst, const float (&z)[4][8],
                                        const float (&k)[4][8], float coef,
                                        int nlo, int half) {
#pragma unroll
    for (int nt = 0; nt < 4; ++nt)
#pragma unroll
        for (int r = 0; r < 8; ++r) {
            float v = z[nt][r] + coef * k[nt][r];
            zst[(r + 8 * half) * DIM + nt * 16 + nlo] = (_Float16)v;
        }
}

// ---- one ODE function eval: k = tanh(zst@W1+b1)@W2+b2 ----------------------
// Biases are read from LDS (b1l/b2l), so the steady-state loop is LDS+WMMA only.
__device__ __forceinline__ void f_eval(const _Float16* zst, _Float16* hst,
                                       const _Float16* w1sw, const _Float16* w2sw,
                                       const float* b1l, const float* b2l,
                                       int lane, int nlo, int half,
                                       float (&k)[4][8]) {
    // GEMM1: [16,64]x[64,256] -> tanh -> stage hidden as f16
    for (int nt = 0; nt < 16; ++nt) {
        float bv = b1l[nt * 16 + nlo];
        v8f c;
#pragma unroll
        for (int r = 0; r < 8; ++r) c[r] = bv;
#pragma unroll
        for (int kt = 0; kt < 2; ++kt) {
            v16h a = load_afrag(zst, DIM, nlo, half, kt);
            v16h b = load_bfrag(w1sw + (kt * 16 + nt) * 512, lane);
            c = __builtin_amdgcn_wmma_f32_16x16x32_f16(false, a, false, b,
                                                       (short)0, c, false, false);
        }
#pragma unroll
        for (int r = 0; r < 8; ++r) {
            float hv = fast_tanh(c[r]);
            hst[(r + 8 * half) * HID + nt * 16 + nlo] = (_Float16)hv;
        }
    }
    // GEMM2: [16,256]x[256,64] -> k (in C/D register layout)
#pragma unroll
    for (int nt = 0; nt < 4; ++nt) {
        float bv = b2l[nt * 16 + nlo];
        v8f c;
#pragma unroll
        for (int r = 0; r < 8; ++r) c[r] = bv;
        for (int kt = 0; kt < 8; ++kt) {
            v16h a = load_afrag(hst, HID, nlo, half, kt);
            v16h b = load_bfrag(w2sw + (kt * 4 + nt) * 512, lane);
            c = __builtin_amdgcn_wmma_f32_16x16x32_f16(false, a, false, b,
                                                       (short)0, c, false, false);
        }
#pragma unroll
        for (int r = 0; r < 8; ++r) k[nt][r] = c[r];
    }
}

__global__ void __launch_bounds__(THREADS)
node_rk4_kernel(const float* __restrict__ z0, const float* __restrict__ t,
                const float* __restrict__ W1, const float* __restrict__ b1,
                const float* __restrict__ W2, const float* __restrict__ b2,
                float* __restrict__ out) {
    extern __shared__ char smem[];
    _Float16* w1sw = (_Float16*)(smem + OFF_W1);
    _Float16* w2sw = (_Float16*)(smem + OFF_W2);
    float*    b1l  = (float*)(smem + OFF_B1);
    float*    b2l  = (float*)(smem + OFF_B2);

    const int tid  = threadIdx.x;
    const int wave = tid >> 5;
    const int lane = tid & 31;
    const int nlo  = lane & 15;
    const int half = lane >> 4;

    _Float16* zst = (_Float16*)(smem + OFF_Z + wave * 2048);
    _Float16* hst = (_Float16*)(smem + OFF_H + wave * 8192);

    // ---- stage weights into LDS in WMMA B-fragment layout (once) ----------
    for (int fi = 0; fi < 4; ++fi) {
        const int F = wave * 4 + fi;
        {   // W1: fragment id F = kt*16 + nt  (kt in [0,2), nt in [0,16))
            const int kt = F >> 4, nt = F & 15;
#pragma unroll
            for (int e = 0; e < 16; ++e) {
                int i = e >> 1;
                int kk = 2 * (i & 3) + 8 * half + 16 * (i >> 2) + 32 * kt + (e & 1);
                int n  = nt * 16 + nlo;
                w1sw[F * 512 + lane * 16 + e] = (_Float16)W1[kk * HID + n];
            }
        }
        {   // W2: fragment id F = kt*4 + nt  (kt in [0,8), nt in [0,4))
            const int kt = F >> 2, nt = F & 3;
#pragma unroll
            for (int e = 0; e < 16; ++e) {
                int i = e >> 1;
                int kk = 2 * (i & 3) + 8 * half + 16 * (i >> 2) + 32 * kt + (e & 1);
                int n  = nt * 16 + nlo;
                w2sw[F * 512 + lane * 16 + e] = (_Float16)W2[kk * DIM + n];
            }
        }
    }
    // biases into LDS (hot loop then never touches global for operands)
    b1l[tid] = b1[tid];
    if (tid < DIM) b2l[tid] = b2[tid];
    __syncthreads();

    const int row0 = blockIdx.x * ROWS_PER_BLOCK + wave * 16;

    // ---- load z0 tile into registers (C/D layout) and emit step-0 output ---
    float z[4][8];
#pragma unroll
    for (int nt = 0; nt < 4; ++nt)
#pragma unroll
        for (int r = 0; r < 8; ++r) {
            int row = row0 + r + 8 * half;
            z[nt][r] = z0[(size_t)row * DIM + nt * 16 + nlo];
            out[(size_t)row * DIM + nt * 16 + nlo] = z[nt][r];
        }

    // ---- RK4 time loop ------------------------------------------------------
    for (int s = 0; s < TLEN - 1; ++s) {
        const float h = t[s + 1] - t[s];
        float kk[4][8] = {};
        float acc[4][8];

        // k1
        stage_z(zst, z, kk, 0.0f, nlo, half);
        f_eval(zst, hst, w1sw, w2sw, b1l, b2l, lane, nlo, half, kk);
#pragma unroll
        for (int nt = 0; nt < 4; ++nt)
#pragma unroll
            for (int r = 0; r < 8; ++r) acc[nt][r] = kk[nt][r];

        // k2
        stage_z(zst, z, kk, 0.5f * h, nlo, half);
        f_eval(zst, hst, w1sw, w2sw, b1l, b2l, lane, nlo, half, kk);
#pragma unroll
        for (int nt = 0; nt < 4; ++nt)
#pragma unroll
            for (int r = 0; r < 8; ++r) acc[nt][r] += 2.0f * kk[nt][r];

        // k3
        stage_z(zst, z, kk, 0.5f * h, nlo, half);
        f_eval(zst, hst, w1sw, w2sw, b1l, b2l, lane, nlo, half, kk);
#pragma unroll
        for (int nt = 0; nt < 4; ++nt)
#pragma unroll
            for (int r = 0; r < 8; ++r) acc[nt][r] += 2.0f * kk[nt][r];

        // k4
        stage_z(zst, z, kk, h, nlo, half);
        f_eval(zst, hst, w1sw, w2sw, b1l, b2l, lane, nlo, half, kk);
#pragma unroll
        for (int nt = 0; nt < 4; ++nt)
#pragma unroll
            for (int r = 0; r < 8; ++r) acc[nt][r] += kk[nt][r];

        // z update + trajectory store (only mandatory HBM traffic)
        const float h6 = h * (1.0f / 6.0f);
        const size_t obase = (size_t)(s + 1) * BSZ * DIM;
#pragma unroll
        for (int nt = 0; nt < 4; ++nt)
#pragma unroll
            for (int r = 0; r < 8; ++r) {
                z[nt][r] += h6 * acc[nt][r];
                int row = row0 + r + 8 * half;
                out[obase + (size_t)row * DIM + nt * 16 + nlo] = z[nt][r];
            }
    }
}

extern "C" void kernel_launch(void* const* d_in, const int* in_sizes, int n_in,
                              void* d_out, int out_size, void* d_ws, size_t ws_size,
                              hipStream_t stream) {
    const float* z0 = (const float*)d_in[0];
    const float* t  = (const float*)d_in[1];
    const float* W1 = (const float*)d_in[2];
    const float* b1 = (const float*)d_in[3];
    const float* W2 = (const float*)d_in[4];
    const float* b2 = (const float*)d_in[5];
    float* out = (float*)d_out;

    // allow >64KB dynamic LDS (~145 KB; two blocks per 320 KB WGP)
    hipFuncSetAttribute((const void*)node_rk4_kernel,
                        hipFuncAttributeMaxDynamicSharedMemorySize, SMEM_BYTES);

    dim3 grid(BSZ / ROWS_PER_BLOCK);   // 512 blocks
    dim3 block(THREADS);               // 8 waves (wave32)
    node_rk4_kernel<<<grid, block, SMEM_BYTES, stream>>>(z0, t, W1, b1, W2, b2, out);
}
